// InterpretableMultiHeadAttention_87454124081232
// MI455X (gfx1250) — matrix operand order repaired and checked by reading, not compiled, not run on hardware
//
#include <hip/hip_runtime.h>

// InterpretableMultiHeadAttention for MI455X (gfx1250, wave32, WMMA).
// B=2, S=2048, H=8, D=128 (per head), d_model=128. fp32 in/out,
// f16 staging + f32 WMMA accumulation for all GEMM-shaped work.

#define B_DIM 2
#define S_DIM 2048
#define H_DIM 8
#define D_DIM 128
#define DM    128              // d_model
#define NHD   (H_DIM * D_DIM)  // 1024
#define LN_EPS 1e-3f

typedef __attribute__((ext_vector_type(16))) _Float16 v16h;
typedef __attribute__((ext_vector_type(8)))  _Float16 v8h;
typedef __attribute__((ext_vector_type(8)))  float    v8f;

__device__ __forceinline__ v8f wmma_f16(v16h a, v16h b, v8f c) {
  // D = A(16x32 f16) * B(32x16 f16) + C(16x16 f32)
  return __builtin_amdgcn_wmma_f32_16x16x32_f16(
      /*neg_a=*/false, a, /*neg_b=*/false, b,
      /*c_mod=*/(short)0, c, /*reuse_a=*/false, /*reuse_b=*/false);
}

// ---------------------------------------------------------------------------
// Kernel 0: weight transpose fp32[rows][cols] -> f16 WT[cols][rows]
// (n-major so WMMA B-fragments read 16 contiguous K values per lane)
// ---------------------------------------------------------------------------
__global__ void transpose_to_f16(const float* __restrict__ W,
                                 _Float16* __restrict__ WT,
                                 int rows, int cols) {
  int idx = blockIdx.x * blockDim.x + threadIdx.x;
  if (idx >= rows * cols) return;
  int r = idx / cols, c = idx - r * cols;
  WT[c * rows + r] = (_Float16)W[r * cols + c];
}

// ---------------------------------------------------------------------------
// Kernel 1: projection GEMM  Y = X[4096,128] @ W[128,1024]
// 8 waves/block; wave = 16x64 strip; K=128 in 4 wmma steps.
// transposed==0 : store [B,H,S,D]  (Q with scale folded, K)
// transposed==1 : store [B,H,D,S]  (V, so it is a ready B-matrix for P*V)
// ---------------------------------------------------------------------------
__global__ __launch_bounds__(256)
void proj_kernel(const float* __restrict__ X,
                 const _Float16* __restrict__ WT,   // [1024][128] n-major
                 _Float16* __restrict__ out,
                 int transposed, float scale) {
  const int lane  = threadIdx.x & 31;
  const int wave  = threadIdx.x >> 5;
  const int half  = lane >> 4;       // 0 | 1
  const int n15   = lane & 15;
  const int m_base = blockIdx.y * 128 + wave * 16;
  const int n_base = blockIdx.x * 64;
  const int mrow   = m_base + n15;

  v8f acc[4] = {};
#pragma unroll
  for (int ks = 0; ks < 4; ++ks) {
    const int k0 = ks * 32;
    // A-fragment: lane=M, 16 contiguous K at (lane>>4)*16
    v16h a;
    const float* xp = X + mrow * DM + k0 + half * 16;
#pragma unroll
    for (int i = 0; i < 8; ++i) a[i]     = (_Float16)xp[i];
#pragma unroll
    for (int i = 0; i < 8; ++i) a[8 + i] = (_Float16)xp[8 + i];
#pragma unroll
    for (int t = 0; t < 4; ++t) {
      // B-fragment: lane=N, 16 contiguous K at (lane>>4)*16
      const int ncol = n_base + t * 16 + n15;
      v16h b = *(const v16h*)(WT + ncol * DM + k0 + half * 16);
      acc[t] = wmma_f16(a, b, acc[t]);
    }
  }
  // scatter into per-head layout (f16)
#pragma unroll
  for (int t = 0; t < 4; ++t) {
    const int n = n_base + t * 16 + n15;
    const int h = n >> 7, d = n & 127;
#pragma unroll
    for (int r = 0; r < 8; ++r) {
      const int m  = m_base + r + half * 8;       // global row in [0,4096)
      const int bb = m >> 11, s = m & (S_DIM - 1);
      const int bh = bb * H_DIM + h;
      const _Float16 val = (_Float16)(acc[t][r] * scale);
      if (transposed)
        out[(bh * D_DIM + d) * S_DIM + s] = val;
      else
        out[(bh * S_DIM + s) * D_DIM + d] = val;
    }
  }
}

// ---------------------------------------------------------------------------
// Kernel 2: flash-style causal attention with MULTIPLICATIVE mask.
// mask[q,k] = (k<=q) ? 1 : 0 applied to scores BEFORE softmax, so masked
// entries contribute exp(0-m)/Z (NOT zero weight) -> process all key blocks.
// 4 waves/block; wave owns 16 queries, iterates keys 32 at a time.
// ---------------------------------------------------------------------------
__global__ __launch_bounds__(128)
void attn_kernel(const _Float16* __restrict__ Qh,   // [BH][S][D], scale folded
                 const _Float16* __restrict__ Kh,   // [BH][S][D]
                 const _Float16* __restrict__ VhT,  // [BH][D][S]
                 float* __restrict__ Ohead) {       // [B*S][H*D] fp32
  __shared__ __align__(64) _Float16 pbuf[4][16][32];   // per-wave P tile
  const int lane = threadIdx.x & 31;
  const int wave = threadIdx.x >> 5;
  const int half = lane >> 4;
  const int n15  = lane & 15;
  const int bh   = blockIdx.y;
  const int q_base = blockIdx.x * 64 + wave * 16;

  const _Float16* Qp = Qh  + (long)bh * S_DIM * D_DIM;
  const _Float16* Kp = Kh  + (long)bh * S_DIM * D_DIM;
  const _Float16* Vp = VhT + (long)bh * D_DIM * S_DIM;

  // Q A-fragments (resident for the whole key loop)
  v16h qf[4];
#pragma unroll
  for (int ks = 0; ks < 4; ++ks) {
    const _Float16* qp = Qp + (q_base + n15) * D_DIM + ks * 32 + half * 16;
    v8h lo = *(const v8h*)qp;
    v8h hi = *(const v8h*)(qp + 8);
#pragma unroll
    for (int i = 0; i < 8; ++i) { qf[ks][i] = lo[i]; qf[ks][8 + i] = hi[i]; }
  }

  float mrun[8], lrun[8];
#pragma unroll
  for (int r = 0; r < 8; ++r) { mrun[r] = -3.0e38f; lrun[r] = 0.f; }
  v8f o[8] = {};

  for (int j = 0; j < S_DIM; j += 32) {
    if (j + 32 < S_DIM) {   // pull next K/V block toward the WGP
      __builtin_prefetch(Kp + (j + 32 + n15) * D_DIM, 0, 1);
      __builtin_prefetch(Vp + n15 * S_DIM + j + 32, 0, 1);
    }
    // scores: two 16x16 tiles covering keys [j, j+32)
    v8f sc0 = {}, sc1 = {};
#pragma unroll
    for (int ks = 0; ks < 4; ++ks) {
      const int dko = ks * 32 + half * 16;
      v16h b0 = *(const v16h*)(Kp + (j + n15) * D_DIM + dko);
      sc0 = wmma_f16(qf[ks], b0, sc0);
      v16h b1 = *(const v16h*)(Kp + (j + 16 + n15) * D_DIM + dko);
      sc1 = wmma_f16(qf[ks], b1, sc1);
    }
    // multiplicative mask + online softmax (row = VGPR r, cols across lanes)
    float p0[8], p1[8], cfac[8];
#pragma unroll
    for (int r = 0; r < 8; ++r) {
      const int qrow = q_base + r + half * 8;
      float s0 = ((j + n15)      <= qrow) ? sc0[r] : 0.f;
      float s1 = ((j + 16 + n15) <= qrow) ? sc1[r] : 0.f;
      float mx = fmaxf(s0, s1);
#pragma unroll
      for (int off = 8; off > 0; off >>= 1) mx = fmaxf(mx, __shfl_xor(mx, off));
      const float mnew = fmaxf(mrun[r], mx);
      const float c  = __expf(mrun[r] - mnew);
      const float e0 = __expf(s0 - mnew);
      const float e1 = __expf(s1 - mnew);
      float sum = e0 + e1;
#pragma unroll
      for (int off = 8; off > 0; off >>= 1) sum += __shfl_xor(sum, off);
      lrun[r] = lrun[r] * c + sum;
      mrun[r] = mnew;
      cfac[r] = c; p0[r] = e0; p1[r] = e1;
    }
#pragma unroll
    for (int f = 0; f < 8; ++f)
#pragma unroll
      for (int r = 0; r < 8; ++r) o[f][r] *= cfac[r];

    // Reshape P (C-layout) -> A-fragment via per-wave LDS bounce
    _Float16* pb = &pbuf[wave][0][0];
#pragma unroll
    for (int r = 0; r < 8; ++r) {
      pb[(r + half * 8) * 32 + n15]      = (_Float16)p0[r];
      pb[(r + half * 8) * 32 + 16 + n15] = (_Float16)p1[r];
    }
    asm volatile("s_wait_dscnt 0" ::: "memory");  // same-wave LDS RAW fence
    v16h pf;
    {
      const _Float16* pl = &pbuf[wave][n15][0] + half * 16;
      v8h lo = *(const v8h*)pl;
      v8h hi = *(const v8h*)(pl + 8);
#pragma unroll
      for (int i = 0; i < 8; ++i) { pf[i] = lo[i]; pf[8 + i] = hi[i]; }
    }
    // O += P(16x32) * V(32x128): V^T layout is a ready B-matrix
#pragma unroll
    for (int f = 0; f < 8; ++f) {
      v16h vb = *(const v16h*)(Vp + (f * 16 + n15) * S_DIM + j + half * 16);
      o[f] = wmma_f16(pf, vb, o[f]);
    }
  }

  // normalize and write [B*S][H*D] fp32 for the output projection
  float inv[8];
#pragma unroll
  for (int r = 0; r < 8; ++r) inv[r] = 1.f / lrun[r];
  const int bb = bh >> 3, h = bh & 7;
#pragma unroll
  for (int f = 0; f < 8; ++f) {
    const int col = h * D_DIM + f * 16 + n15;
#pragma unroll
    for (int r = 0; r < 8; ++r) {
      const int qrow = q_base + r + half * 8;
      Ohead[(long)(bb * S_DIM + qrow) * NHD + col] = o[f][r] * inv[r];
    }
  }
}

// ---------------------------------------------------------------------------
// Kernel 3: Y = Ohead[4096,1024] @ Wo[1024,128], then LayerNorm(axis=-1).
// 8 waves/block: block owns a full 16x128 row-slab -> LN fused via LDS.
// ---------------------------------------------------------------------------
__global__ __launch_bounds__(256)
void outproj_ln_kernel(const float* __restrict__ Ohead,
                       const _Float16* __restrict__ WoT,  // [128][1024] n-major
                       const float* __restrict__ gamma,
                       const float* __restrict__ beta,
                       float* __restrict__ out) {          // [4096][128]
  __shared__ float obuf[16][128];
  __shared__ float mu_s[16], rs_s[16];
  const int lane = threadIdx.x & 31;
  const int wave = threadIdx.x >> 5;
  const int half = lane >> 4;
  const int n15  = lane & 15;
  const int m_base = blockIdx.x * 16;
  const int mrow   = m_base + n15;

  v8f acc = {};
  for (int ks = 0; ks < 32; ++ks) {
    const int k0 = ks * 32;
    v16h a;
    const float* xp = Ohead + (long)mrow * NHD + k0 + half * 16;
#pragma unroll
    for (int i = 0; i < 8; ++i) a[i]     = (_Float16)xp[i];
#pragma unroll
    for (int i = 0; i < 8; ++i) a[8 + i] = (_Float16)xp[8 + i];
    v16h b = *(const v16h*)(WoT + (wave * 16 + n15) * NHD + k0 + half * 16);
    acc = wmma_f16(a, b, acc);
  }
#pragma unroll
  for (int r = 0; r < 8; ++r)
    obuf[r + half * 8][wave * 16 + n15] = acc[r];
  __syncthreads();

  if (threadIdx.x < 16) {
    float s = 0.f, ss = 0.f;
#pragma unroll 8
    for (int c = 0; c < 128; ++c) {
      float v = obuf[threadIdx.x][c];
      s += v; ss += v * v;
    }
    const float mu  = s  * (1.f / 128.f);
    const float var = ss * (1.f / 128.f) - mu * mu;   // matches jnp.var
    mu_s[threadIdx.x] = mu;
    rs_s[threadIdx.x] = rsqrtf(var + LN_EPS);
  }
  __syncthreads();

#pragma unroll
  for (int i = 0; i < 8; ++i) {
    const int e = threadIdx.x + i * 256;
    const int r = e >> 7, c = e & 127;
    out[(long)(m_base + r) * DM + c] =
        (obuf[r][c] - mu_s[r]) * rs_s[r] * gamma[c] + beta[c];
  }
}

// ---------------------------------------------------------------------------
extern "C" void kernel_launch(void* const* d_in, const int* in_sizes, int n_in,
                              void* d_out, int out_size, void* d_ws, size_t ws_size,
                              hipStream_t stream) {
  const float* q     = (const float*)d_in[0];
  const float* k     = (const float*)d_in[1];
  const float* v     = (const float*)d_in[2];
  // d_in[3] is the tril mask; it is exactly (k<=q), computed in-kernel.
  const float* Wq    = (const float*)d_in[4];
  const float* Wk    = (const float*)d_in[5];
  const float* Wv    = (const float*)d_in[6];
  const float* Wo    = (const float*)d_in[7];
  const float* gamma = (const float*)d_in[8];
  const float* beta  = (const float*)d_in[9];
  float* out = (float*)d_out;

  // workspace carve-up (~41 MB total)
  char* ws = (char*)d_ws;
  const size_t wT   = (size_t)NHD * DM * sizeof(_Float16);               // 256 KB
  const size_t hBuf = (size_t)B_DIM * H_DIM * S_DIM * D_DIM * sizeof(_Float16); // 8 MB
  _Float16* WqT = (_Float16*)ws;  ws += wT;
  _Float16* WkT = (_Float16*)ws;  ws += wT;
  _Float16* WvT = (_Float16*)ws;  ws += wT;
  _Float16* WoT = (_Float16*)ws;  ws += wT;
  _Float16* Qh  = (_Float16*)ws;  ws += hBuf;
  _Float16* Kh  = (_Float16*)ws;  ws += hBuf;
  _Float16* VhT = (_Float16*)ws;  ws += hBuf;
  float*    Ohead = (float*)ws;   // 16 MB: [4096][1024] fp32

  const float scale = 0.088388347648318447f;  // 1/sqrt(128)

  // 1) weight transposes (fp32 -> f16, output-channel-major)
  {
    dim3 g((DM * NHD + 255) / 256);
    transpose_to_f16<<<g, 256, 0, stream>>>(Wq, WqT, DM, NHD);
    transpose_to_f16<<<g, 256, 0, stream>>>(Wk, WkT, DM, NHD);
    transpose_to_f16<<<g, 256, 0, stream>>>(Wv, WvT, DM, NHD);
    transpose_to_f16<<<g, 256, 0, stream>>>(Wo, WoT, NHD, DM);
  }
  // 2) QKV projections (scale folded into Q)
  {
    dim3 g(NHD / 64, (B_DIM * S_DIM) / 128);   // 16 x 32
    proj_kernel<<<g, 256, 0, stream>>>(q, WqT, Qh, 0, scale);
    proj_kernel<<<g, 256, 0, stream>>>(k, WkT, Kh, 0, 1.0f);
    proj_kernel<<<g, 256, 0, stream>>>(v, WvT, VhT, 1, 1.0f);
  }
  // 3) attention
  {
    dim3 g(S_DIM / 64, B_DIM * H_DIM);         // 32 x 16
    attn_kernel<<<g, 128, 0, stream>>>(Qh, Kh, VhT, Ohead);
  }
  // 4) output projection + LayerNorm
  {
    dim3 g((B_DIM * S_DIM) / 16);              // 256
    outproj_ln_kernel<<<g, 256, 0, stream>>>(Ohead, WoT, gamma, beta, out);
  }
  (void)in_sizes; (void)n_in; (void)out_size; (void)ws_size;
}
